// MACBlock_77189152244394
// MI455X (gfx1250) — matrix operand order repaired and checked by reading, not compiled, not run on hardware
//
#include <hip/hip_runtime.h>
#include <hip/hip_bf16.h>
#include <math.h>

// ---------------------------------------------------------------------------
// MAC transformer block for MI455X (gfx1250, wave32, WMMA bf16 path).
// All GEMMs run on v_wmma_f32_16x16x32_bf16 with f32 accumulation and
// software-pipelined (double-buffered) fragment loads.
// Weights are repacked per call (deterministic) into B-fragment layout.
// ---------------------------------------------------------------------------

typedef __bf16 bf16_t;
typedef __attribute__((ext_vector_type(16))) __bf16 v16bf;
typedef __attribute__((ext_vector_type(8)))  __bf16 v8bf;
typedef __attribute__((ext_vector_type(8)))  float  v8f;

#define B_     2
#define N_     2048
#define DIM_   1024
#define H_     16
#define DH_    64
#define SEG_   512
#define P_     32          // NPM + NM prefix keys
#define KV_    (N_ + P_)   // 2080
#define DFF_   2730
#define DFFP_  2752        // ff2 K padded to multiple of 32
#define NPAD1_ 2816        // swiglu half-N padded to 11*256
#define EPS_   1.1920929e-07f
#define MROWS_ (B_ * N_)   // 4096

static __device__ __forceinline__ v8f wmma_bf16(v16bf a, v16bf b, v8f c) {
  return __builtin_amdgcn_wmma_f32_16x16x32_bf16(false, a, false, b,
                                                 (short)0, c, false, false);
}

// A-matrix 16x32 bf16 fragment: lanes 0-15 hold K in
// [0,16), lanes 16-31 hold [16,32) -> two 16B loads per lane.
static __device__ __forceinline__ v16bf load_a16(const bf16_t* row, int lh) {
  union { v16bf v; v8bf h[2]; } u;
  u.h[0] = *(const v8bf*)(row + lh * 16);
  u.h[1] = *(const v8bf*)(row + lh * 16 + 8);
  return u.v;
}

// ---------------------------------------------------------------------------
// Weight packing: f32 (Kin x Nin, leading dim ldw, column offset coloff) ->
// bf16 B-fragment tiles. Element (k,n) -> ((n/16)*Kt + k/32)*512
//                                         + (n%16)*32 + (k%32).
// Lane L then loads its B fragment as one contiguous 32B chunk.
// ---------------------------------------------------------------------------
__global__ void __launch_bounds__(256)
pack_b_kernel(const float* __restrict__ W, bf16_t* __restrict__ Pk,
              int Kin, int Nin, int ldw, int coloff, int Kt) {
  const size_t idx   = (size_t)blockIdx.x * 256 + threadIdx.x;
  const size_t chunk = idx >> 9;
  const int within   = (int)(idx & 511);
  const int t = (int)(chunk % (size_t)Kt);
  const int u = (int)(chunk / (size_t)Kt);
  const int n = u * 16 + (within >> 5);
  const int k = t * 32 + (within & 31);
  float v = 0.f;
  if (k < Kin && n < Nin) v = W[(size_t)k * ldw + coloff + n];
  Pk[idx] = (bf16_t)v;
}

// ---------------------------------------------------------------------------
// Generic WMMA GEMM: out(f32, M x Nld) = A(bf16, M x K) * Bpacked
//                    (+ bias[n]) (+ resid[m,n]).
// Block: 256 threads = 8 waves, tile 64x256; wave tile 32x64.
// Double-buffered A and B fragments: loads for step t+1 are in flight while
// the 8 WMMAs of step t execute (avoids s_wait_loadcnt 0 before every WMMA).
// ---------------------------------------------------------------------------
__global__ void __launch_bounds__(256)
gemm_f32out_kernel(const bf16_t* __restrict__ A, const bf16_t* __restrict__ Bp,
                   const float* __restrict__ bias,
                   const float* __restrict__ resid,
                   float* __restrict__ out, int K, int Nld) {
  const int lane = threadIdx.x & 31, wave = threadIdx.x >> 5;
  const int l16 = lane & 15, lh = lane >> 4;
  const int Kt = K >> 5;
  const int mbase = blockIdx.y * 64 + (wave >> 2) * 32;
  const int nbase = blockIdx.x * 256 + (wave & 3) * 64;
  v8f acc[2][4] = {};
  const bf16_t* a0p = A + (size_t)(mbase + l16) * K;
  const bf16_t* a1p = a0p + (size_t)16 * K;
  const bf16_t* bp0 = Bp + (size_t)(nbase >> 4) * Kt * 512 + l16 * 32 + lh * 16;

  v16bf a0c = load_a16(a0p, lh);
  v16bf a1c = load_a16(a1p, lh);
  v16bf bc[4];
#pragma unroll
  for (int nt = 0; nt < 4; ++nt)
    bc[nt] = *(const v16bf*)(bp0 + (size_t)nt * Kt * 512);

#pragma unroll 2
  for (int t = 0; t < Kt - 1; ++t) {
    // issue next-step fragment loads first
    v16bf a0n = load_a16(a0p + (t + 1) * 32, lh);
    v16bf a1n = load_a16(a1p + (t + 1) * 32, lh);
    v16bf bn[4];
#pragma unroll
    for (int nt = 0; nt < 4; ++nt)
      bn[nt] = *(const v16bf*)(bp0 + ((size_t)nt * Kt + t + 1) * 512);
    __builtin_prefetch((const void*)(a0p + (t + 16) * 32), 0, 1);
    __builtin_prefetch((const void*)(bp0 + (size_t)(t + 16) * 512), 0, 1);
    // compute on current buffers (only waits for the previous load group)
#pragma unroll
    for (int nt = 0; nt < 4; ++nt) {
      acc[0][nt] = wmma_bf16(a0c, bc[nt], acc[0][nt]);
      acc[1][nt] = wmma_bf16(a1c, bc[nt], acc[1][nt]);
    }
    a0c = a0n; a1c = a1n;
#pragma unroll
    for (int nt = 0; nt < 4; ++nt) bc[nt] = bn[nt];
  }
#pragma unroll
  for (int nt = 0; nt < 4; ++nt) {
    acc[0][nt] = wmma_bf16(a0c, bc[nt], acc[0][nt]);
    acc[1][nt] = wmma_bf16(a1c, bc[nt], acc[1][nt]);
  }

#pragma unroll
  for (int mt = 0; mt < 2; ++mt)
#pragma unroll
    for (int nt = 0; nt < 4; ++nt) {
      const int n = nbase + nt * 16 + l16;
      if (n < Nld) {
#pragma unroll
        for (int r = 0; r < 8; ++r) {
          const size_t m = (size_t)mbase + mt * 16 + r + 8 * lh;
          float v = acc[mt][nt][r];
          if (bias)  v += bias[n];
          if (resid) v += resid[m * Nld + n];
          out[m * (size_t)Nld + n] = v;
        }
      }
    }
}

// ---------------------------------------------------------------------------
// Fused FF1 + SwiGLU: dual-B GEMM (a-half / g-half of ff_w1), epilogue
// writes bf16 silu(g+bg)*(a+ba) into hg (ld = DFFP_, zero pad to 2752).
// 128 accumulator VGPRs -> double-buffer A only; all 8 B fragments for a
// step are issued as one group (single wait point per K-step).
// ---------------------------------------------------------------------------
__global__ void __launch_bounds__(256)
gemm_swiglu_kernel(const bf16_t* __restrict__ A, const bf16_t* __restrict__ Bpa,
                   const bf16_t* __restrict__ Bpg, const float* __restrict__ b1,
                   bf16_t* __restrict__ hg, int K) {
  const int lane = threadIdx.x & 31, wave = threadIdx.x >> 5;
  const int l16 = lane & 15, lh = lane >> 4;
  const int Kt = K >> 5;
  const int mbase = blockIdx.y * 64 + (wave >> 2) * 32;
  const int nbase = blockIdx.x * 256 + (wave & 3) * 64;
  v8f aca[2][4] = {}, acg[2][4] = {};
  const bf16_t* a0p = A + (size_t)(mbase + l16) * K;
  const bf16_t* a1p = a0p + (size_t)16 * K;
  const size_t bofs = (size_t)(nbase >> 4) * Kt * 512 + l16 * 32 + lh * 16;
  const bf16_t* bpa = Bpa + bofs;
  const bf16_t* bpg = Bpg + bofs;

  v16bf a0c = load_a16(a0p, lh);
  v16bf a1c = load_a16(a1p, lh);
  for (int t = 0; t < Kt; ++t) {
    // all 8 B fragments for this step in one load group
    v16bf ba[4], bg[4];
#pragma unroll
    for (int nt = 0; nt < 4; ++nt) {
      const size_t o = ((size_t)nt * Kt + t) * 512;
      ba[nt] = *(const v16bf*)(bpa + o);
      bg[nt] = *(const v16bf*)(bpg + o);
    }
    // next-step A in flight during the WMMA burst
    v16bf a0n, a1n;
    if (t + 1 < Kt) {
      a0n = load_a16(a0p + (t + 1) * 32, lh);
      a1n = load_a16(a1p + (t + 1) * 32, lh);
    }
    __builtin_prefetch((const void*)(bpa + (size_t)(t + 8) * 512), 0, 1);
    __builtin_prefetch((const void*)(bpg + (size_t)(t + 8) * 512), 0, 1);
#pragma unroll
    for (int nt = 0; nt < 4; ++nt) {
      aca[0][nt] = wmma_bf16(a0c, ba[nt], aca[0][nt]);
      aca[1][nt] = wmma_bf16(a1c, ba[nt], aca[1][nt]);
      acg[0][nt] = wmma_bf16(a0c, bg[nt], acg[0][nt]);
      acg[1][nt] = wmma_bf16(a1c, bg[nt], acg[1][nt]);
    }
    if (t + 1 < Kt) { a0c = a0n; a1c = a1n; }
  }

#pragma unroll
  for (int mt = 0; mt < 2; ++mt)
#pragma unroll
    for (int nt = 0; nt < 4; ++nt) {
      const int n = nbase + nt * 16 + l16;
      if (n < DFFP_) {
#pragma unroll
        for (int r = 0; r < 8; ++r) {
          const size_t m = (size_t)mbase + mt * 16 + r + 8 * lh;
          float val = 0.f;
          if (n < DFF_) {
            const float av = aca[mt][nt][r] + b1[n];
            const float gv = acg[mt][nt][r] + b1[DFF_ + n];
            val = av * (gv / (1.f + __expf(-gv)));
          }
          hg[m * (size_t)DFFP_ + n] = (bf16_t)val;
        }
      }
    }
}

// ---------------------------------------------------------------------------
// Flash attention over prefix (32 keys) + same-segment causal window.
// Grid (32 qblocks, 16 heads, 2 batch), 128 threads = 4 waves x 16 q-rows.
// Q/K row-major bf16 [bh][pos][64]; V transposed [bh][64][2080].
// ---------------------------------------------------------------------------
__global__ void __launch_bounds__(128)
attn_kernel(const bf16_t* __restrict__ Q, const bf16_t* __restrict__ KF,
            const bf16_t* __restrict__ VT, bf16_t* __restrict__ aout) {
  const int qb = blockIdx.x, h = blockIdx.y, b = blockIdx.z;
  const int wave = threadIdx.x >> 5, lane = threadIdx.x & 31;
  const int l16 = lane & 15, lh = lane >> 4;
  const int qbase = qb * 64 + wave * 16;
  const size_t hb = (size_t)(b * H_ + h);
  const bf16_t* Qh = Q + hb * (size_t)N_ * DH_;
  const bf16_t* Kh = KF + hb * (size_t)KV_ * DH_;
  const bf16_t* Vh = VT + hb * (size_t)DH_ * KV_;
  __shared__ bf16_t plds[4 * 16 * 32];
  bf16_t* pl = plds + wave * 512;

  const bf16_t* qrow = Qh + (size_t)(qbase + l16) * DH_;
  v16bf qa0 = load_a16(qrow, lh);
  v16bf qa1 = load_a16(qrow + 32, lh);
  v8f o0 = {}, o1 = {}, o2 = {}, o3 = {};
  float mrow[8], lrow[8];
#pragma unroll
  for (int r = 0; r < 8; ++r) { mrow[r] = -3.0e38f; lrow[r] = 0.f; }

  const int seg = qbase >> 9;
  const int segbase = P_ + seg * SEG_;
  const int qoff = qbase - seg * SEG_;
  const int nseg = (qoff + 47) >> 5;  // ceil((qoff+16)/32) 32-key blocks
  for (int it = 0; it <= nseg; ++it) {
    const int jbase = (it == 0) ? 0 : segbase + (it - 1) * 32;
    // prefetch next key block (K rows + V columns) into cache
    if (it < nseg) {
      const int jn = (it == 0) ? segbase : jbase + 32;
      __builtin_prefetch((const void*)(Kh + (size_t)(jn + l16) * DH_), 0, 1);
      __builtin_prefetch((const void*)(Vh + (size_t)l16 * KV_ + jn), 0, 1);
      __builtin_prefetch((const void*)(Vh + (size_t)(16 + l16) * KV_ + jn), 0, 1);
    }
    // scores: q(16x64) @ k^T -> two 16x16 tiles
    const bf16_t* krow0 = Kh + (size_t)(jbase + l16) * DH_;
    const bf16_t* krow1 = krow0 + 16 * DH_;
    v16bf kb00 = *(const v16bf*)(krow0 + lh * 16);
    v16bf kb01 = *(const v16bf*)(krow0 + 32 + lh * 16);
    v16bf kb10 = *(const v16bf*)(krow1 + lh * 16);
    v16bf kb11 = *(const v16bf*)(krow1 + 32 + lh * 16);
    v8f s0 = {}, s1 = {};
    s0 = wmma_bf16(qa0, kb00, s0);
    s0 = wmma_bf16(qa1, kb01, s0);
    s1 = wmma_bf16(qa0, kb10, s1);
    s1 = wmma_bf16(qa1, kb11, s1);
#pragma unroll
    for (int r = 0; r < 8; ++r) {
      float x0 = s0[r] * 0.125f, x1 = s1[r] * 0.125f;  // 1/sqrt(64)
      if (it > 0) {
        const int qi = qbase + r + 8 * lh;
        const int j0 = jbase - P_ + l16;
        if (j0 > qi)      x0 = -3.0e38f;
        if (j0 + 16 > qi) x1 = -3.0e38f;
      }
      float mx = fmaxf(x0, x1);
#pragma unroll
      for (int d = 1; d < 16; d <<= 1) mx = fmaxf(mx, __shfl_xor(mx, d, 32));
      const float mnew = fmaxf(mrow[r], mx);
      const float corr = __expf(mrow[r] - mnew);
      mrow[r] = mnew;
      const float e0 = __expf(x0 - mnew);
      const float e1 = __expf(x1 - mnew);
      float rs = e0 + e1;
#pragma unroll
      for (int d = 1; d < 16; d <<= 1) rs += __shfl_xor(rs, d, 32);
      lrow[r] = lrow[r] * corr + rs;
      o0[r] *= corr; o1[r] *= corr; o2[r] *= corr; o3[r] *= corr;
      const int prow = (r + 8 * lh) * 32;       // C-layout -> LDS 16x32 tile
      pl[prow + l16]      = (bf16_t)e0;
      pl[prow + 16 + l16] = (bf16_t)e1;
    }
    asm volatile("s_wait_dscnt 0" ::: "memory");
    v16bf pa = load_a16(pl + l16 * 32, lh);     // probs in A-fragment layout
    const bf16_t* vcol = Vh + (size_t)l16 * KV_ + jbase + lh * 16;
    v16bf vb0 = *(const v16bf*)(vcol);
    v16bf vb1 = *(const v16bf*)(vcol + 16 * KV_);
    v16bf vb2 = *(const v16bf*)(vcol + 32 * KV_);
    v16bf vb3 = *(const v16bf*)(vcol + 48 * KV_);
    o0 = wmma_bf16(pa, vb0, o0);
    o1 = wmma_bf16(pa, vb1, o1);
    o2 = wmma_bf16(pa, vb2, o2);
    o3 = wmma_bf16(pa, vb3, o3);
  }
#pragma unroll
  for (int r = 0; r < 8; ++r) {
    const float invl = 1.0f / lrow[r];
    const size_t m = (size_t)b * N_ + qbase + r + 8 * lh;
    bf16_t* orow = aout + m * (size_t)DIM_ + h * DH_ + l16;
    orow[0]  = (bf16_t)(o0[r] * invl);
    orow[16] = (bf16_t)(o1[r] * invl);
    orow[32] = (bf16_t)(o2[r] * invl);
    orow[48] = (bf16_t)(o3[r] * invl);
  }
}

// ---------------------------------------------------------------------------
// RMSNorm over DIM_=1024, bf16 output. One block per row.
// ---------------------------------------------------------------------------
__global__ void __launch_bounds__(256)
rmsnorm_kernel(const float* __restrict__ x, const float* __restrict__ w,
               bf16_t* __restrict__ out) {
  const int row = blockIdx.x, tid = threadIdx.x;
  const float* xr = x + (size_t)row * DIM_;
  float v[4]; float ss = 0.f;
#pragma unroll
  for (int i = 0; i < 4; ++i) { v[i] = xr[tid + i * 256]; ss += v[i] * v[i]; }
  __shared__ float red[256];
  red[tid] = ss; __syncthreads();
  for (int s = 128; s > 0; s >>= 1) {
    if (tid < s) red[tid] += red[tid + s];
    __syncthreads();
  }
  const float scale = rsqrtf(red[0] * (1.0f / DIM_) + EPS_);
  bf16_t* orow = out + (size_t)row * DIM_;
#pragma unroll
  for (int i = 0; i < 4; ++i)
    orow[tid + i * 256] = (bf16_t)(v[i] * scale * w[tid + i * 256]);
}

// mean over sequence then RMSNorm -> pooled[b][d], f32
__global__ void __launch_bounds__(1024)
pool_kernel(const float* __restrict__ mem_out, const float* __restrict__ w,
            float* __restrict__ pooled) {
  const int b = blockIdx.x, d = threadIdx.x;
  const float* p = mem_out + (size_t)b * N_ * DIM_ + d;
  float s = 0.f;
  for (int i = 0; i < N_; ++i) s += p[(size_t)i * DIM_];
  const float mean = s * (1.0f / N_);
  __shared__ float red[1024];
  red[d] = mean * mean; __syncthreads();
  for (int t = 512; t > 0; t >>= 1) {
    if (d < t) red[d] += red[d + t];
    __syncthreads();
  }
  const float scale = rsqrtf(red[0] * (1.0f / DIM_) + EPS_);
  pooled[b * DIM_ + d] = mean * scale * w[d];
}

// out[b][o] = dot(vecs[b][:1024], W[:, o]) with leading dim ldw
__global__ void __launch_bounds__(256)
matvec_kernel(const float* __restrict__ vecs, const float* __restrict__ W,
              int ldw, float* __restrict__ out) {
  const int o = blockIdx.x * 256 + threadIdx.x;
  const int b = blockIdx.y;
  const float* v = vecs + (size_t)b * DIM_;
  float acc = 0.f;
  for (int k = 0; k < DIM_; ++k) acc += v[k] * W[(size_t)k * ldw + o];
  out[(size_t)b * DIM_ + o] = acc;
}

// Fill prefix rows [0,32) of KF (row-major) and VT (transposed):
// rows 0..15 = k_extra/v_extra (identical mem-token rows), 16..31 = persist.
__global__ void __launch_bounds__(256)
prefix_kernel(const float* __restrict__ ke, const float* __restrict__ ve,
              const float* __restrict__ pm, bf16_t* __restrict__ KF,
              bf16_t* __restrict__ VT) {
  const int h = blockIdx.x, b = blockIdx.y;
  const size_t hb = (size_t)(b * H_ + h);
  for (int idx = threadIdx.x; idx < 32 * DH_; idx += 256) {
    const int row = idx >> 6, d = idx & 63;
    float kv, vv;
    if (row < 16) {
      kv = ke[(size_t)b * DIM_ + h * DH_ + d];
      vv = ve[(size_t)b * DIM_ + h * DH_ + d];
    } else {
      const float p = pm[((size_t)h * 16 + (row - 16)) * DH_ + d];
      kv = p; vv = p;
    }
    KF[(hb * KV_ + row) * DH_ + d] = (bf16_t)kv;
    VT[(hb * DH_ + d) * KV_ + row] = (bf16_t)vv;
  }
}

// RoPE on q,k; scatter q->Q, k->KF rows [32,2080), v->VT (transposed).
__global__ void __launch_bounds__(512)
rope_kernel(const float* __restrict__ qkv, bf16_t* __restrict__ Q,
            bf16_t* __restrict__ KF, bf16_t* __restrict__ VT) {
  const int i = blockIdx.x;            // b*2048 + n
  const int b = i >> 11, n = i & 2047;
  const int h = threadIdx.x >> 5, p = threadIdx.x & 31;
  const float* r = qkv + (size_t)i * (3 * DIM_) + h * DH_ + 2 * p;
  const float q0 = r[0],        q1 = r[1];
  const float k0 = r[DIM_],     k1 = r[DIM_ + 1];
  const float v0 = r[2 * DIM_], v1 = r[2 * DIM_ + 1];
  const float inv = __powf(10000.0f, -(float)p * (1.0f / 32.0f));
  float sn, cs;
  __sincosf((float)n * inv, &sn, &cs);
  const float qa = q0 * cs - q1 * sn, qb = q1 * cs + q0 * sn;
  const float ka = k0 * cs - k1 * sn, kb = k1 * cs + k0 * sn;
  const size_t hb = (size_t)(b * H_ + h);
  bf16_t* qd = Q + (hb * N_ + n) * DH_ + 2 * p;
  qd[0] = (bf16_t)qa; qd[1] = (bf16_t)qb;
  bf16_t* kd = KF + (hb * KV_ + P_ + n) * DH_ + 2 * p;
  kd[0] = (bf16_t)ka; kd[1] = (bf16_t)kb;
  VT[(hb * DH_ + 2 * p) * KV_ + P_ + n]     = (bf16_t)v0;
  VT[(hb * DH_ + 2 * p + 1) * KV_ + P_ + n] = (bf16_t)v1;
}

// ---------------------------------------------------------------------------
extern "C" void kernel_launch(void* const* d_in, const int* in_sizes, int n_in,
                              void* d_out, int out_size, void* d_ws,
                              size_t ws_size, hipStream_t stream) {
  (void)in_sizes; (void)n_in; (void)out_size; (void)ws_size;
  const float* x        = (const float*)d_in[0];
  const float* mem_out  = (const float*)d_in[1];
  const float* attn_nw  = (const float*)d_in[2];
  const float* qkvw     = (const float*)d_in[3];
  const float* outw     = (const float*)d_in[4];
  const float* pm       = (const float*)d_in[5];
  const float* pool_nw  = (const float*)d_in[6];
  const float* memtok_w = (const float*)d_in[7];
  const float* ff_nw    = (const float*)d_in[8];
  const float* ffw1     = (const float*)d_in[9];
  const float* ffb1     = (const float*)d_in[10];
  const float* ffw2     = (const float*)d_in[11];
  const float* ffb2     = (const float*)d_in[12];
  float* outp = (float*)d_out;

  char* base = (char*)d_ws;
  size_t off = 0;
  auto alloc = [&](size_t bytes) -> char* {
    char* p = base + off;
    off = (off + bytes + 255) & ~(size_t)255;
    return p;
  };
  bf16_t* xn     = (bf16_t*)alloc((size_t)MROWS_ * DIM_ * 2);
  bf16_t* xn2    = (bf16_t*)alloc((size_t)MROWS_ * DIM_ * 2);
  float*  qkvbuf = (float*) alloc((size_t)MROWS_ * 3 * DIM_ * 4);
  bf16_t* Qb     = (bf16_t*)alloc((size_t)B_ * H_ * N_ * DH_ * 2);
  bf16_t* KFb    = (bf16_t*)alloc((size_t)B_ * H_ * KV_ * DH_ * 2);
  bf16_t* VTb    = (bf16_t*)alloc((size_t)B_ * H_ * DH_ * KV_ * 2);
  bf16_t* aout   = (bf16_t*)alloc((size_t)MROWS_ * DIM_ * 2);
  float*  x2     = (float*) alloc((size_t)MROWS_ * DIM_ * 4);
  bf16_t* hg     = (bf16_t*)alloc((size_t)MROWS_ * DFFP_ * 2);
  float*  pooled = (float*) alloc((size_t)B_ * DIM_ * 4);
  float*  mt     = (float*) alloc((size_t)B_ * DIM_ * 4);
  float*  ke     = (float*) alloc((size_t)B_ * DIM_ * 4);
  float*  ve     = (float*) alloc((size_t)B_ * DIM_ * 4);
  bf16_t* pk_qkv = (bf16_t*)alloc((size_t)DIM_ * 3072 * 2);
  bf16_t* pk_out = (bf16_t*)alloc((size_t)DIM_ * 1024 * 2);
  bf16_t* pk_w1a = (bf16_t*)alloc((size_t)DIM_ * NPAD1_ * 2);
  bf16_t* pk_w1g = (bf16_t*)alloc((size_t)DIM_ * NPAD1_ * 2);
  bf16_t* pk_w2  = (bf16_t*)alloc((size_t)DFFP_ * 1024 * 2);

  // --- weight repack (bf16, B-fragment layout) ---
  pack_b_kernel<<<(DIM_ * 3072) / 256, 256, 0, stream>>>(
      qkvw, pk_qkv, DIM_, 3072, 3072, 0, DIM_ / 32);
  pack_b_kernel<<<(DIM_ * 1024) / 256, 256, 0, stream>>>(
      outw, pk_out, DIM_, 1024, 1024, 0, DIM_ / 32);
  pack_b_kernel<<<(DIM_ * NPAD1_) / 256, 256, 0, stream>>>(
      ffw1, pk_w1a, DIM_, DFF_, 2 * DFF_, 0, DIM_ / 32);
  pack_b_kernel<<<(DIM_ * NPAD1_) / 256, 256, 0, stream>>>(
      ffw1, pk_w1g, DIM_, DFF_, 2 * DFF_, DFF_, DIM_ / 32);
  pack_b_kernel<<<(DFFP_ * 1024) / 256, 256, 0, stream>>>(
      ffw2, pk_w2, DFF_, 1024, 1024, 0, DFFP_ / 32);

  // --- mem-token path (tiny) ---
  pool_kernel<<<B_, 1024, 0, stream>>>(mem_out, pool_nw, pooled);
  matvec_kernel<<<dim3(4, B_), 256, 0, stream>>>(pooled, memtok_w, DIM_, mt);
  matvec_kernel<<<dim3(4, B_), 256, 0, stream>>>(mt, qkvw + DIM_, 3072, ke);
  matvec_kernel<<<dim3(4, B_), 256, 0, stream>>>(mt, qkvw + 2 * DIM_, 3072, ve);
  prefix_kernel<<<dim3(H_, B_), 256, 0, stream>>>(ke, ve, pm, KFb, VTb);

  // --- attention ---
  rmsnorm_kernel<<<MROWS_, 256, 0, stream>>>(x, attn_nw, xn);
  gemm_f32out_kernel<<<dim3(12, 64), 256, 0, stream>>>(
      xn, pk_qkv, nullptr, nullptr, qkvbuf, DIM_, 3072);
  rope_kernel<<<MROWS_, 512, 0, stream>>>(qkvbuf, Qb, KFb, VTb);
  attn_kernel<<<dim3(N_ / 64, H_, B_), 128, 0, stream>>>(Qb, KFb, VTb, aout);
  gemm_f32out_kernel<<<dim3(4, 64), 256, 0, stream>>>(
      aout, pk_out, nullptr, x, x2, DIM_, 1024);

  // --- FFN ---
  rmsnorm_kernel<<<MROWS_, 256, 0, stream>>>(x2, ff_nw, xn2);
  gemm_swiglu_kernel<<<dim3(11, 64), 256, 0, stream>>>(
      xn2, pk_w1a, pk_w1g, ffb1, hg, DIM_);
  gemm_f32out_kernel<<<dim3(4, 64), 256, 0, stream>>>(
      hg, pk_w2, ffb2, x2, outp, DFFP_, 1024);
}
